// SAUTE_62749472195354
// MI455X (gfx1250) — matrix operand order repaired and checked by reading, not compiled, not run on hardware
//
#include <hip/hip_runtime.h>
#include <hip/hip_bf16.h>

// Problem constants (match reference setup_inputs)
#define B_ 4
#define T_ 64
#define L_ 128
#define D_ 768
#define H_ 12
#define d_ 64
#define LDSPITCH 776   // 768 + 8 bf16 pad: row stride 1552B -> 4-bank skew, conflict-free

typedef __attribute__((ext_vector_type(16))) __bf16 v16bf;
typedef __attribute__((ext_vector_type(8)))  float  v8f;

// ---------------------------------------------------------------------------
// K0: convert Wq (f32 [D,D]) to bf16 once; L2-resident, reused by all blocks.
// ---------------------------------------------------------------------------
__global__ void __launch_bounds__(256)
cvt_wq_bf16(const float* __restrict__ in, __bf16* __restrict__ out) {
    int i = blockIdx.x * 256 + threadIdx.x;   // n = D*D = 589824
    out[i] = (__bf16)in[i];
}

// ---------------------------------------------------------------------------
// K1: k = edu @ Wk^T, v = edu @ Wv^T.  256 rows x 768 cols, f32.  ~0.6 GFLOP —
// negligible; plain VALU dot products, operands L2-resident.
// ---------------------------------------------------------------------------
__global__ void __launch_bounds__(256)
kv_proj(const float* __restrict__ edu, const float* __restrict__ Wk,
        const float* __restrict__ Wv, float* __restrict__ kout,
        float* __restrict__ vout) {
    int idx = blockIdx.x * 256 + threadIdx.x;         // < 256*768
    int row = idx / D_;
    int col = idx - row * D_;
    const float* x  = edu + (size_t)row * D_;
    const float* wk = Wk  + (size_t)col * D_;
    const float* wv = Wv  + (size_t)col * D_;
    float ka = 0.f, va = 0.f;
    #pragma unroll 8
    for (int i = 0; i < D_; ++i) {
        float xv = x[i];
        ka = fmaf(xv, wk[i], ka);
        va = fmaf(xv, wv[i], va);
    }
    kout[idx] = ka;
    vout[idx] = va;
}

// ---------------------------------------------------------------------------
// K2: speaker-segmented inclusive prefix sum of per-head kv outer products.
// One thread per (b,h,di,dj); scans t=0..63 with 8 per-speaker register
// accumulators (predicated adds, no dynamic array indexing -> no scratch).
// M stored COLUMN-major bf16 (Mcm[dj*64+di]) so the fused kernel's B-operand
// is a contiguous 16x bf16 (32B) load per lane.
// ---------------------------------------------------------------------------
__global__ void __launch_bounds__(256)
seg_prefix(const int* __restrict__ spk, const float* __restrict__ K,
           const float* __restrict__ V, __bf16* __restrict__ Mcm) {
    int idx = blockIdx.x * 256 + threadIdx.x;         // < B*H*d*d = 196608
    int dj = idx & 63;
    int di = (idx >> 6) & 63;
    int bh = idx >> 12;                               // 0..47
    int h  = bh % H_;
    int b  = bh / H_;
    float a0=0,a1=0,a2=0,a3=0,a4=0,a5=0,a6=0,a7=0;
    const int* sp = spk + b * T_;
    for (int t = 0; t < T_; ++t) {
        int bt = b * T_ + t;
        float kv = K[(size_t)bt * D_ + h * d_ + di] *
                   V[(size_t)bt * D_ + h * d_ + dj];
        int s = sp[t];
        a0 += (s == 0) ? kv : 0.f;  a1 += (s == 1) ? kv : 0.f;
        a2 += (s == 2) ? kv : 0.f;  a3 += (s == 3) ? kv : 0.f;
        a4 += (s == 4) ? kv : 0.f;  a5 += (s == 5) ? kv : 0.f;
        a6 += (s == 6) ? kv : 0.f;  a7 += (s == 7) ? kv : 0.f;
        float val = (s < 4) ? ((s < 2) ? (s == 0 ? a0 : a1)
                                       : (s == 2 ? a2 : a3))
                            : ((s < 6) ? (s == 4 ? a4 : a5)
                                       : (s == 6 ? a6 : a7));
        Mcm[((size_t)bt * H_ + h) * (d_ * d_) + dj * d_ + di] = (__bf16)val;
    }
}

// ---------------------------------------------------------------------------
// K3 (fused): out = X + (X @ Wq^T) @ M, per (b,t,16-row tile).
// One block (8 waves) per 16-row strip:
//   phase 0: cooperatively stage X[16x768] -> LDS bf16 (24KB, padded rows)
//   phase 1: each wave computes 6 of 48 q column-tiles (A via ds_load,
//            B = L2-resident Wq bf16), q written straight to a 2nd LDS strip
//            -> the 100MB Q tensor never touches HBM
//   phase 2: each wave computes 6 of 48 out tiles (q @ M, K=64), fused
//            residual add from X (L2/L0-hot), f32 store.
// LDS: 2 x 16*776*2B = 48.5KB << 320KB/WGP.
// ---------------------------------------------------------------------------
__global__ void __launch_bounds__(256)
fused_qm_wmma(const float* __restrict__ X, const __bf16* __restrict__ Wqb,
              const __bf16* __restrict__ Mcm, float* __restrict__ Out) {
    __shared__ __bf16 xs[16 * LDSPITCH];
    __shared__ __bf16 qs[16 * LDSPITCH];

    const int tid  = threadIdx.x;
    const int lane = tid & 31;
    const int wave = tid >> 5;
    const int bt   = blockIdx.x >> 3;                 // 0..255
    const int rt   = blockIdx.x & 7;                  // row tile within L
    const size_t R0 = (size_t)bt * L_ + rt * 16;      // global row base

    // ---- phase 0: stage X rows -> LDS bf16 ----
    {
        const float4* src = (const float4*)(X + R0 * D_);   // 16B-aligned
        for (int c = tid; c < 16 * (D_ / 4); c += 256) {    // 3072 float4
            int row  = c / (D_ / 4);
            int col4 = c - row * (D_ / 4);
            float4 v = src[row * (D_ / 4) + col4];
            int o = row * LDSPITCH + col4 * 4;
            xs[o + 0] = (__bf16)v.x;
            xs[o + 1] = (__bf16)v.y;
            xs[o + 2] = (__bf16)v.z;
            xs[o + 3] = (__bf16)v.w;
        }
    }
    __syncthreads();

    const int m    = lane & 15;      // A row / B col / C col
    const int half = lane >> 4;
    const int ka   = half * 8;       // unused (A loads use kb, contiguous k)
    const int kb   = half * 16;      // A/B K-base (contiguous k = 16*half + j)

    const v8f vzero = {};
    v8f acc[6];

    // ---- phase 1: q strip, 6 column tiles per wave ----
    #pragma unroll
    for (int i = 0; i < 6; ++i) acc[i] = vzero;
    {
        const __bf16* xrow = xs + m * LDSPITCH;
        for (int k0 = 0; k0 < D_; k0 += 32) {
            v16bf a;
            #pragma unroll
            for (int j = 0; j < 16; ++j) {
                a[j]     = xrow[k0 + kb + j];            // ds_load, contiguous k = 16*half + j
            }
            #pragma unroll
            for (int i = 0; i < 6; ++i) {
                const int c0 = (wave * 6 + i) * 16;
                const __bf16* wrow = Wqb + (size_t)(c0 + m) * D_ + kb + k0;
                v16bf b;
                #pragma unroll
                for (int j = 0; j < 16; ++j) b[j] = wrow[j];
                acc[i] = __builtin_amdgcn_wmma_f32_16x16x32_bf16(
                             false, a, false, b, (short)0, acc[i], false, false);
            }
        }
        // C/D layout: VGPR r -> row half*8+r, col m; write q (bf16) to LDS
        #pragma unroll
        for (int i = 0; i < 6; ++i) {
            const int c0 = (wave * 6 + i) * 16;
            #pragma unroll
            for (int r = 0; r < 8; ++r)
                qs[(half * 8 + r) * LDSPITCH + c0 + m] = (__bf16)acc[i][r];
        }
    }
    __syncthreads();

    // ---- phase 2: out tiles = q @ M + residual, 6 tiles per wave ----
    const int bthb = bt * H_;
    #pragma unroll
    for (int i = 0; i < 6; ++i) {
        const int ti  = wave * 6 + i;                 // 0..47
        const int h   = ti >> 2;                      // head
        const int ctm = ti & 3;                       // 16-col tile within d
        const __bf16* qrow = qs + m * LDSPITCH + h * d_;                // A(m,k)
        const __bf16* mb   = Mcm + (size_t)(bthb + h) * (d_ * d_)       // B(k,n)
                                 + (ctm * 16 + m) * d_ + kb;            // col-major M
        v8f c = vzero;
        #pragma unroll
        for (int k0 = 0; k0 < d_; k0 += 32) {
            v16bf a, b;
            #pragma unroll
            for (int j = 0; j < 16; ++j) {
                a[j]     = qrow[k0 + kb + j];
            }
            #pragma unroll
            for (int j = 0; j < 16; ++j) b[j] = mb[k0 + j];
            c = __builtin_amdgcn_wmma_f32_16x16x32_bf16(
                    false, a, false, b, (short)0, c, false, false);
        }
        #pragma unroll
        for (int r = 0; r < 8; ++r) {
            size_t idx = (R0 + half * 8 + r) * D_ + h * d_ + ctm * 16 + m;
            Out[idx] = X[idx] + c[r];                 // residual add (X L2-hot)
        }
    }
}

// ---------------------------------------------------------------------------
extern "C" void kernel_launch(void* const* d_in, const int* in_sizes, int n_in,
                              void* d_out, int out_size, void* d_ws, size_t ws_size,
                              hipStream_t stream) {
    (void)in_sizes; (void)n_in; (void)out_size; (void)ws_size;
    // inputs: input_ids, speaker_ids, token_embeddings, edu_embeddings, Wk, Wv, Wq
    const int*   spk = (const int*)  d_in[1];
    const float* tok = (const float*)d_in[2];
    const float* edu = (const float*)d_in[3];
    const float* Wk  = (const float*)d_in[4];
    const float* Wv  = (const float*)d_in[5];
    const float* Wq  = (const float*)d_in[6];
    float* out = (float*)d_out;

    // workspace carve-up (all 256B aligned)
    char* ws = (char*)d_ws;
    __bf16* Wqb  = (__bf16*)(ws);                               // 768*768*2   = 1179648
    float*  kbuf = (float*) (ws + 1179648);                     // 256*768*4   = 786432
    float*  vbuf = (float*) (ws + 1966080);                     // 256*768*4   = 786432
    __bf16* Mcm  = (__bf16*)(ws + 2752512);                     // 3072*4096*2 = 25165824
                                                                // total ~27.9 MB

    cvt_wq_bf16  <<<(D_*D_)/256, 256, 0, stream>>>(Wq, Wqb);
    kv_proj      <<<(B_*T_*D_)/256, 256, 0, stream>>>(edu, Wk, Wv, kbuf, vbuf);
    seg_prefix   <<<(B_*H_*d_*d_)/256, 256, 0, stream>>>(spk, kbuf, vbuf, Mcm);
    fused_qm_wmma<<<B_*T_*(L_/16), 256, 0, stream>>>(tok, Wqb, Mcm, out);  // 2048 blocks
}
